// VQVAE2_57578331571011
// MI455X (gfx1250) — compile-verified, matrix operand order checked by reading, not run on hardware
//
#include <hip/hip_runtime.h>

// ---------------------------------------------------------------------------
// VQ-VAE codebook lookup + EMA update for MI455X (gfx1250, wave32, WMMA).
// Sizes fixed by the reference: B=16, D=64, T=4096, E=1024.
// Codebook staged through LDS with async global->LDS double buffering
// (ASYNCcnt-tracked) so WMMA overlaps the copies; ||e||^2 table in LDS.
// ---------------------------------------------------------------------------

#define VQ_B 16
#define VQ_D 64
#define VQ_T 4096
#define VQ_E 1024
#define MTILE 128            // t-rows per workgroup (8 waves x 16 rows)
#define NCHUNK (VQ_E / 16)   // 64 chunks of 16 codewords
#define GCHUNK 4             // chunks per staged group
#define NGROUP (NCHUNK / GCHUNK)          // 16 groups
#define GELEMS (GCHUNK * 2 * 32 * 16)     // bf16 elems per group (hi or lo) = 4096

typedef __attribute__((ext_vector_type(16))) __bf16 v16bf;
typedef __attribute__((ext_vector_type(8)))  __bf16 v8bf;
typedef __attribute__((ext_vector_type(8)))  float  v8f;
typedef __attribute__((ext_vector_type(4)))  int    v4i;

#if __has_builtin(__builtin_amdgcn_global_load_async_to_lds_b128)
#define HAVE_ASYNC 1
#else
#define HAVE_ASYNC 0
#endif

#if HAVE_ASYNC
typedef __attribute__((address_space(1))) v4i* gptr128_t;
typedef __attribute__((address_space(3))) v4i* lptr128_t;
#endif

__device__ __forceinline__ void copy16_lds(__bf16* ldst, const __bf16* gsrc) {
#if HAVE_ASYNC
  __builtin_amdgcn_global_load_async_to_lds_b128(
      (gptr128_t)(gsrc),          // global src (addrspace(1) int4*)
      (lptr128_t)(ldst),          // LDS dst    (addrspace(3) int4*)
      /*offset=*/0, /*cpol=*/0);
#else
  *(v8bf*)ldst = *(const v8bf*)gsrc;   // sync fallback: 16B global->reg->LDS
#endif
}

template <int N>
__device__ __forceinline__ void wait_async() {
#if HAVE_ASYNC
#if __has_builtin(__builtin_amdgcn_s_wait_asynccnt)
  __builtin_amdgcn_s_wait_asynccnt(N);
#else
  asm volatile("s_wait_asynccnt %0" ::"n"(N) : "memory");
#endif
#endif
}

__device__ __forceinline__ v8f wmma_bf16(v16bf a, v16bf b, v8f c) {
  // D = A(16x32 bf16) x B(32x16 bf16) + C(16x16 f32)
  return __builtin_amdgcn_wmma_f32_16x16x32_bf16(
      /*neg_a=*/false, a, /*neg_b=*/false, b,
      /*c_mod=*/(short)0, c, /*reuse_a=*/false, /*reuse_b=*/false);
}

// A/B 16-bit operand element -> K mapping from ISA 7.12.2:
// lane half h = lane>>4 ; element i in [0,16):
//   k = kstep*32 + (i<8 ? 8*h + i : 16 + 8*h + (i-8))
__device__ __forceinline__ int wmma16_k(int s, int half, int i) {
  return s * 32 + ((i < 8) ? (half * 8 + i) : (16 + half * 8 + (i - 8)));
}

// ---------------------------------------------------------------------------
// 0) zero accumulators (counts + embed_sum) in workspace
// ---------------------------------------------------------------------------
__global__ void vq_zero(float* __restrict__ p, int n) {
  int g = blockIdx.x * blockDim.x + threadIdx.x;
  if (g < n) p[g] = 0.0f;
}

// ---------------------------------------------------------------------------
// 1) prep: split embedding into bf16 hi/lo pre-swizzled B-operand layout,
//    flat index = ((c*2 + s)*32 + lane)*16 + i  ; also ||e||^2.
// ---------------------------------------------------------------------------
__global__ void vq_prep(const float* __restrict__ emb,   // [E, D]
                        __bf16* __restrict__ ebuf_hi,
                        __bf16* __restrict__ ebuf_lo,
                        float* __restrict__ e2) {
  int gid = blockIdx.x * blockDim.x + threadIdx.x;   // 65536 total
  if (gid < NCHUNK * 2 * 32 * 16) {
    int i    = gid & 15;
    int lane = (gid >> 4) & 31;
    int s    = (gid >> 9) & 1;
    int c    = gid >> 10;
    int n    = c * 16 + (lane & 15);
    int half = lane >> 4;
    int k    = wmma16_k(s, half, i);
    float v  = emb[n * VQ_D + k];
    __bf16 h = (__bf16)v;
    ebuf_hi[gid] = h;
    ebuf_lo[gid] = (__bf16)(v - (float)h);
  }
  if (gid < VQ_E) {
    const float* r = emb + gid * VQ_D;
    float s2 = 0.0f;
#pragma unroll
    for (int k = 0; k < VQ_D; ++k) s2 += r[k] * r[k];
    e2[gid] = s2;
  }
}

// ---------------------------------------------------------------------------
// 2) main: distances via bf16x3 WMMA with async-LDS staged codebook,
//    argmin, gather outputs, scatter EMA stats.
//    grid = B * (T/MTILE) = 512 blocks, 256 threads (8 wave32)
// ---------------------------------------------------------------------------
__global__ void __launch_bounds__(256)
vq_main(const float* __restrict__ x,        // [B, D, T]
        const float* __restrict__ emb,      // [E, D]
        const __bf16* __restrict__ ebuf_hi,
        const __bf16* __restrict__ ebuf_lo,
        const float* __restrict__ e2,       // [E]
        float* __restrict__ o_embed_idx,    // [B, T, D]
        float* __restrict__ o_qx,           // [B, D, T]
        float* __restrict__ o_qidx,         // [B, T] (as float)
        float* __restrict__ counts,         // [E]
        float* __restrict__ embed_sum) {    // [D, E]
  __shared__ __bf16 lds_hi[2][GELEMS];      // 2 x 8 KB
  __shared__ __bf16 lds_lo[2][GELEMS];      // 2 x 8 KB
  __shared__ float  lds_e2[VQ_E];           // 4 KB
  __shared__ int    s_idx[8][16];

  const int b    = blockIdx.x >> 5;            // T/MTILE = 32 tiles per batch
  const int t0   = (blockIdx.x & 31) * MTILE;
  const int wave = threadIdx.x >> 5;
  const int lane = threadIdx.x & 31;
  const int half = lane >> 4;
  const int tw   = t0 + wave * 16;             // this wave's 16 rows
  const int tA   = tw + (lane & 15);           // A-operand row for this lane

  const float* xb = x + (size_t)b * VQ_D * VQ_T;

  // ---- preload ||e||^2 table into LDS (visibility via first barrier) ----
  for (int j = threadIdx.x; j < VQ_E; j += 256) lds_e2[j] = e2[j];

  // ---- issue async copy of codebook group 0 ----
  {
    const __bf16* sh = ebuf_hi;
    const __bf16* sl = ebuf_lo;
#pragma unroll
    for (int j = 0; j < 2; ++j) {
      int off = (threadIdx.x + j * 256) * 8;   // 16B per thread-slot
      copy16_lds(&lds_hi[0][off], sh + off);
      copy16_lds(&lds_lo[0][off], sl + off);
    }
  }

  // ---- load A tile (16 rows x K=64) in WMMA layout, coalesced over t ----
  float a[32];
#pragma unroll
  for (int s = 0; s < 2; ++s)
#pragma unroll
    for (int i = 0; i < 16; ++i) {
      int k = wmma16_k(s, half, i);
      a[s * 16 + i] = xb[(size_t)k * VQ_T + tA];
    }

  // bf16 hi/lo split (error-compensated GEMM)
  v16bf ah[2], al[2];
#pragma unroll
  for (int s = 0; s < 2; ++s)
#pragma unroll
    for (int i = 0; i < 16; ++i) {
      float v = a[s * 16 + i];
      __bf16 h = (__bf16)v;
      ah[s][i] = h;
      al[s][i] = (__bf16)(v - (float)h);
    }

  float best[8];
  int   bidx[8];
#pragma unroll
  for (int r = 0; r < 8; ++r) { best[r] = 3.4e38f; bidx[r] = 0; }

  // ---- sweep codebook: 16 double-buffered groups of 4 chunks ----
  for (int g = 0; g < NGROUP; ++g) {
    // all waves are done computing from the buffer we are about to overwrite
    __syncthreads();
    if (g + 1 < NGROUP) {
      const __bf16* sh = ebuf_hi + (size_t)(g + 1) * GELEMS;
      const __bf16* sl = ebuf_lo + (size_t)(g + 1) * GELEMS;
      __bf16* dh = lds_hi[(g + 1) & 1];
      __bf16* dl = lds_lo[(g + 1) & 1];
#pragma unroll
      for (int j = 0; j < 2; ++j) {
        int off = (threadIdx.x + j * 256) * 8;
        copy16_lds(&dh[off], sh + off);
        copy16_lds(&dl[off], sl + off);
      }
      wait_async<4>();   // own group-g copies done; g+1's 4 still in flight
    } else {
      wait_async<0>();   // last group: drain everything
    }
    __syncthreads();     // every wave's group-g copies have landed in LDS

    const __bf16* bufh = lds_hi[g & 1];
    const __bf16* bufl = lds_lo[g & 1];

    for (int cc = 0; cc < GCHUNK; ++cc) {
      const int c = g * GCHUNK + cc;
      v16bf bh0 = *(const v16bf*)(bufh + ((cc * 2 + 0) * 32 + lane) * 16);
      v16bf bh1 = *(const v16bf*)(bufh + ((cc * 2 + 1) * 32 + lane) * 16);
      v16bf bl0 = *(const v16bf*)(bufl + ((cc * 2 + 0) * 32 + lane) * 16);
      v16bf bl1 = *(const v16bf*)(bufl + ((cc * 2 + 1) * 32 + lane) * 16);

      v8f acc_hh = {};   // hi*hi
      v8f acc_x  = {};   // cross terms
      acc_hh = wmma_bf16(ah[0], bh0, acc_hh);
      acc_x  = wmma_bf16(ah[0], bl0, acc_x);
      acc_x  = wmma_bf16(al[0], bh0, acc_x);
      acc_hh = wmma_bf16(ah[1], bh1, acc_hh);
      acc_x  = wmma_bf16(ah[1], bl1, acc_x);
      acc_x  = wmma_bf16(al[1], bh1, acc_x);

      const int n  = c * 16 + (lane & 15);   // this lane's codeword column
      const float t2 = lds_e2[n];
#pragma unroll
      for (int r = 0; r < 8; ++r) {
        // dist(m,n) up to +||x||^2 (constant per row, irrelevant to argmin)
        float sc = fmaf(-2.0f, acc_hh[r] + acc_x[r], t2);
        if (sc < best[r]) { best[r] = sc; bidx[r] = n; }
      }
    }
  }

  // ---- argmin reduction across the 16 lanes holding the same rows ----
#pragma unroll
  for (int off = 8; off >= 1; off >>= 1) {
#pragma unroll
    for (int r = 0; r < 8; ++r) {
      float ob = __shfl_xor(best[r], off, 32);
      int   oi = __shfl_xor(bidx[r], off, 32);
      if (ob < best[r] || (ob == best[r] && oi < bidx[r])) {
        best[r] = ob; bidx[r] = oi;
      }
    }
  }

  // rows 0..7 live in lanes 0..15 (half 0), rows 8..15 in lanes 16..31
  if ((lane & 15) == 0) {
#pragma unroll
    for (int r = 0; r < 8; ++r) s_idx[wave][half * 8 + r] = bidx[r];
  }
  __syncthreads();

  // ---- quantized_idx (stored as float) ----
  if (lane < 16) o_qidx[(size_t)b * VQ_T + tw + lane] = (float)s_idx[wave][lane];

  // ---- embed_idx [B,T,D]: gather codeword rows, float2-coalesced ----
#pragma unroll 4
  for (int m = 0; m < 16; ++m) {
    const float2* src = (const float2*)(emb + (size_t)s_idx[wave][m] * VQ_D);
    float2* dst = (float2*)(o_embed_idx + ((size_t)(b * VQ_T + tw + m)) * VQ_D);
    dst[lane] = src[lane];
  }

  // ---- embed_idx_qx [B,D,T] (numerically identical, transposed layout) ----
  {
    const int m  = lane & 15;
    const int em = s_idx[wave][m];
#pragma unroll
    for (int j = 0; j < 32; ++j) {
      int d = j + 32 * half;
      o_qx[((size_t)(b * VQ_D + d)) * VQ_T + tw + m] = emb[(size_t)em * VQ_D + d];
    }
  }

  // ---- histogram ----
  if (lane < 16) atomicAdd(&counts[s_idx[wave][lane]], 1.0f);

  // ---- embed_sum[d][e] scatter-add, reusing the A registers ----
  {
    const int em = s_idx[wave][lane & 15];
#pragma unroll
    for (int s = 0; s < 2; ++s)
#pragma unroll
      for (int i = 0; i < 16; ++i) {
        int k = wmma16_k(s, half, i);
        atomicAdd(&embed_sum[(size_t)k * VQ_E + em], a[s * 16 + i]);
      }
  }
}

// ---------------------------------------------------------------------------
// 3) EMA size update + global normalization (single 1024-thread WG)
// ---------------------------------------------------------------------------
__global__ void __launch_bounds__(1024)
vq_fin1(const float* __restrict__ ema_size, const float* __restrict__ counts,
        float* __restrict__ out_sizes) {
  __shared__ float red[1024];
  const int e = threadIdx.x;
  float raw = 0.99f * ema_size[e] + 0.01f * counts[e];
  red[e] = raw;
  __syncthreads();
  for (int off = 512; off >= 1; off >>= 1) {
    if (e < off) red[e] += red[e + off];
    __syncthreads();
  }
  const float nsum = red[0];
  out_sizes[e] = (raw + 1e-5f) / (nsum + (float)VQ_E * 1e-5f) * nsum;
}

// ---------------------------------------------------------------------------
// 4) EMA weight update + new (transposed) embedding
// ---------------------------------------------------------------------------
__global__ void vq_fin2(const float* __restrict__ ema_w,      // [D, E]
                        const float* __restrict__ embed_sum,  // [D, E]
                        const float* __restrict__ sizes,      // [E] (normalized)
                        float* __restrict__ out_ema_w,        // [D, E]
                        float* __restrict__ out_embedding) {  // [E, D]
  int gid = blockIdx.x * blockDim.x + threadIdx.x;            // D*E = 65536
  if (gid >= VQ_D * VQ_E) return;
  int d = gid >> 10, e = gid & (VQ_E - 1);
  float w = 0.99f * ema_w[gid] + 0.01f * embed_sum[gid];
  out_ema_w[gid] = w;
  out_embedding[(size_t)e * VQ_D + d] = w / sizes[e];
}

// ---------------------------------------------------------------------------
extern "C" void kernel_launch(void* const* d_in, const int* in_sizes, int n_in,
                              void* d_out, int out_size, void* d_ws, size_t ws_size,
                              hipStream_t stream) {
  const float* x        = (const float*)d_in[0];   // [B, D, T]
  const float* emb      = (const float*)d_in[1];   // [E, D]
  const float* ema_size = (const float*)d_in[2];   // [E]
  const float* ema_w    = (const float*)d_in[3];   // [D, E]

  float* out = (float*)d_out;
  float* o_embed_idx = out;                                    // 4,194,304
  float* o_qx        = out + 4194304;                          // 4,194,304
  float* o_qidx      = out + 8388608;                          //    65,536
  float* o_embedding = out + 8454144;                          //    65,536
  float* o_sizes     = out + 8519680;                          //     1,024
  float* o_emaw      = out + 8520704;                          //    65,536

  // workspace layout (~0.52 MB)
  char* ws = (char*)d_ws;
  __bf16* ebuf_hi   = (__bf16*)(ws);                 // 131072 B
  __bf16* ebuf_lo   = (__bf16*)(ws + 131072);        // 131072 B
  float*  e2        = (float*)(ws + 262144);         //   4096 B
  float*  counts    = (float*)(ws + 266240);         //   4096 B
  float*  embed_sum = (float*)(ws + 270336);         // 262144 B

  // counts + embed_sum are contiguous: zero both in one launch
  vq_zero<<<(66560 + 255) / 256, 256, 0, stream>>>(counts, 66560);
  vq_prep<<<256, 256, 0, stream>>>(emb, ebuf_hi, ebuf_lo, e2);
  vq_main<<<VQ_B * (VQ_T / MTILE), 256, 0, stream>>>(
      x, emb, ebuf_hi, ebuf_lo, e2,
      o_embed_idx, o_qx, o_qidx, counts, embed_sum);
  vq_fin1<<<1, 1024, 0, stream>>>(ema_size, counts, o_sizes);
  vq_fin2<<<(VQ_D * VQ_E + 255) / 256, 256, 0, stream>>>(
      ema_w, embed_sum, o_sizes, o_emaw, o_embedding);
}
